// GPT2MoEAttention_81509889344105
// MI455X (gfx1250) — compile-verified, matrix operand order checked by reading, not run on hardware
//
#include <hip/hip_runtime.h>

typedef __bf16 bf16;
typedef __attribute__((ext_vector_type(16))) __bf16 v16bf;
typedef __attribute__((ext_vector_type(8)))  __bf16 v8bf;
typedef __attribute__((ext_vector_type(4)))  __bf16 v4bf;
typedef __attribute__((ext_vector_type(8)))  float  v8f;

// s_wait_asynccnt <= n (ASYNCcnt tracks global_load_async_to_lds_*)
#define S_WAIT_ASYNC(nstr) asm volatile("s_wait_asynccnt " nstr ::: "memory")

// CDNA5 async copy: global memory -> LDS, 16 bytes per lane, no VGPR data.
__device__ __forceinline__ void async_b128(void* lds, const void* g) {
    unsigned int off = (unsigned int)(unsigned long long)lds;
    asm volatile("global_load_async_to_lds_b128 %0, %1, off"
                 :: "v"(off), "v"(g) : "memory");
}

// Butterfly max across a 16-lane row via DPP ROW_XMASK (VALU only, no DScnt).
template <int CTRL>
__device__ __forceinline__ float fmax_xor_dpp(float x) {
    int yi = __builtin_amdgcn_update_dpp(0, __float_as_int(x), CTRL, 0xf, 0xf, true);
    return fmaxf(x, __int_as_float(yi));
}

// ---------------------------------------------------------------------------
// Fragment loader: 16x32 (16-bit) WMMA operand layout.
//   lane <  16: row = lane,    K = {k0..k0+7,  k0+16..k0+23}
//   lane >= 16: row = lane-16, K = {k0+8..+15, k0+24..+31}
// ---------------------------------------------------------------------------
__device__ __forceinline__ v16bf load_frag(const bf16* p, int ld, int lane) {
    const int r    = lane & 15;
    const int ksel = (lane >> 4) << 3;
    union { v16bf v; v8bf h[2]; } u;
    u.h[0] = *(const v8bf*)(p + r * ld + ksel);
    u.h[1] = *(const v8bf*)(p + r * ld + ksel + 16);
    return u.v;
}

// ---------------------------------------------------------------------------
__global__ __launch_bounds__(256) void cvt4_f32_bf16(const float* __restrict__ in,
                                                     bf16* __restrict__ out) {
    int i = blockIdx.x * 256 + threadIdx.x;
    float4 f = ((const float4*)in)[i];
    v4bf o;
    o[0] = (bf16)f.x; o[1] = (bf16)f.y; o[2] = (bf16)f.z; o[3] = (bf16)f.w;
    ((v4bf*)out)[i] = o;
}

// fp32 W[K,N] -> bf16 Wt[N,K]
__global__ __launch_bounds__(256) void cvt_transpose(const float* __restrict__ W,
                                                     bf16* __restrict__ Wt,
                                                     int K, int N) {
    __shared__ float tile[32][33];
    const int n0 = blockIdx.x * 32, k0 = blockIdx.y * 32;
    const int t = threadIdx.x;
    {
        int kk = t >> 3, nn = (t & 7) << 2;
        float4 v = *(const float4*)&W[(size_t)(k0 + kk) * N + n0 + nn];
        tile[kk][nn + 0] = v.x; tile[kk][nn + 1] = v.y;
        tile[kk][nn + 2] = v.z; tile[kk][nn + 3] = v.w;
    }
    __syncthreads();
    {
        int nn = t >> 3, kk = (t & 7) << 2;
        v4bf o;
        o[0] = (bf16)tile[kk + 0][nn]; o[1] = (bf16)tile[kk + 1][nn];
        o[2] = (bf16)tile[kk + 2][nn]; o[3] = (bf16)tile[kk + 3][nn];
        *(v4bf*)&Wt[(size_t)(n0 + nn) * K + k0 + kk] = o;
    }
}

// ---------------------------------------------------------------------------
// bf16 GEMM: A[M,1024] @ Wt[N,1024]^T + bias.
// Block tile 256x128, BK=32, 8 waves (4x2), wave tile 64x64 (16 WMMA/k-step).
// Double-buffered async-to-LDS staging (ASYNCcnt).
// MODE 0: scatter q/k -> [B,H,S,64], v -> [B,H,64,S] bf16 ; MODE 1: fp32 out
// ---------------------------------------------------------------------------
template <int MODE>
__global__ __launch_bounds__(256) void gemm_bf16(const bf16* __restrict__ A,
                                                 const bf16* __restrict__ Bt,
                                                 const float* __restrict__ bias,
                                                 int Ndim,
                                                 bf16* __restrict__ qb,
                                                 bf16* __restrict__ kb,
                                                 bf16* __restrict__ vTb,
                                                 float* __restrict__ out) {
    constexpr int Kdim = 1024;
    constexpr int NIT  = Kdim / 32;
    __shared__ bf16 As[2][256][40];
    __shared__ bf16 Bs[2][128][40];

    const int tid   = threadIdx.x;
    const int lane  = tid & 31;
    const int w     = tid >> 5;
    const int waveM = w >> 1;     // 0..3 -> 64 rows each
    const int waveN = w & 1;      // 0..1 -> 64 cols each
    const int m0    = blockIdx.y * 256;
    const int n0    = blockIdx.x * 128;

    // staging: A row = tid (4 x b128), B row = tid>>1 (2 x b128)
    const int br = tid >> 1;
    const int bc = (tid & 1) << 4;
    const bf16* arow = A  + (size_t)(m0 + tid) * Kdim;
    const bf16* brow = Bt + (size_t)(n0 + br) * Kdim + bc;

    v8f acc[4][4] = {};

    async_b128(&As[0][tid][0],  arow);
    async_b128(&As[0][tid][8],  arow + 8);
    async_b128(&As[0][tid][16], arow + 16);
    async_b128(&As[0][tid][24], arow + 24);
    async_b128(&Bs[0][br][bc],     brow);
    async_b128(&Bs[0][br][bc + 8], brow + 8);

    for (int it = 0; it < NIT; ++it) {
        const int buf = it & 1;
        if (it + 1 < NIT) {
            const int nb = buf ^ 1, kt = (it + 1) * 32;
            async_b128(&As[nb][tid][0],  arow + kt);
            async_b128(&As[nb][tid][8],  arow + kt + 8);
            async_b128(&As[nb][tid][16], arow + kt + 16);
            async_b128(&As[nb][tid][24], arow + kt + 24);
            async_b128(&Bs[nb][br][bc],     brow + kt);
            async_b128(&Bs[nb][br][bc + 8], brow + kt + 8);
            S_WAIT_ASYNC("0x6");   // current buffer's 6 loads done (in-order)
        } else {
            S_WAIT_ASYNC("0x0");
        }
        __syncthreads();

        v16bf af[4], bfr[4];
#pragma unroll
        for (int mi = 0; mi < 4; ++mi)
            af[mi] = load_frag(&As[buf][waveM * 64 + mi * 16][0], 40, lane);
#pragma unroll
        for (int ni = 0; ni < 4; ++ni)
            bfr[ni] = load_frag(&Bs[buf][waveN * 64 + ni * 16][0], 40, lane);
#pragma unroll
        for (int mi = 0; mi < 4; ++mi)
#pragma unroll
            for (int ni = 0; ni < 4; ++ni)
                acc[mi][ni] = __builtin_amdgcn_wmma_f32_16x16x32_bf16(
                    false, af[mi], false, bfr[ni], (short)0, acc[mi][ni], false, false);
        __syncthreads();
    }

    // Epilogue: D layout -> N = lane&15, M = r + 8*(lane>>4)
    const int nlane = lane & 15;
    const int mofs  = (lane >> 4) << 3;
#pragma unroll
    for (int mi = 0; mi < 4; ++mi) {
#pragma unroll
        for (int ni = 0; ni < 4; ++ni) {
            int n   = n0 + waveN * 64 + ni * 16 + nlane;
            float bv = bias[n];
            if constexpr (MODE == 0) {
                int part = n >> 10, c = n & 1023;
                int h = c >> 6, e = c & 63;
#pragma unroll
                for (int r = 0; r < 8; ++r) {
                    int m = m0 + waveM * 64 + mi * 16 + mofs + r;
                    int b = m >> 11, s = m & 2047;
                    bf16 ov = (bf16)(acc[mi][ni][r] + bv);
                    if (part == 0)
                        qb[(((size_t)(b * 16 + h) * 2048) + s) * 64 + e] = ov;
                    else if (part == 1)
                        kb[(((size_t)(b * 16 + h) * 2048) + s) * 64 + e] = ov;
                    else
                        vTb[(((size_t)(b * 16 + h) * 64) + e) * 2048 + s] = ov;
                }
            } else {
#pragma unroll
                for (int r = 0; r < 8; ++r) {
                    int m = m0 + waveM * 64 + mi * 16 + mofs + r;
                    out[(size_t)m * Ndim + n] = acc[mi][ni][r] + bv;
                }
            }
        }
    }
}

// ---------------------------------------------------------------------------
// Causal flash attention. grid = (S/64, B*H), 128 threads (4 waves).
// Row-max via DPP xor-butterfly; row-sum via P @ ones WMMA (no ds_bpermute).
// ---------------------------------------------------------------------------
__global__ __launch_bounds__(128) void attn_kernel(const bf16* __restrict__ qb,
                                                   const bf16* __restrict__ kb,
                                                   const bf16* __restrict__ vTb,
                                                   bf16* __restrict__ ob) {
    const int bh   = blockIdx.y;        // b*16 + h
    const int qt   = blockIdx.x * 64;
    const int tid  = threadIdx.x;
    const int lane = tid & 31;
    const int w    = tid >> 5;          // 0..3

    __shared__ bf16 Ks[2][32][72];      // [key][hd]
    __shared__ bf16 Vs[2][64][40];      // [hd][key]
    __shared__ bf16 Ps[4][16][40];      // per-wave P tile

    const bf16* qhead = qb + (size_t)bh * 2048 * 64;
    const bf16* khead = kb + (size_t)bh * 2048 * 64;
    const bf16* vhead = vTb + (size_t)bh * 64 * 2048;

    v16bf qf[2];
    qf[0] = load_frag(qhead + (size_t)(qt + w * 16) * 64, 64, lane);
    qf[1] = load_frag(qhead + (size_t)(qt + w * 16) * 64 + 32, 64, lane);

    v16bf onesf;
#pragma unroll
    for (int i = 0; i < 16; ++i) onesf[i] = (bf16)1.0f;

    float mrow[8];
    v8f oacc[4] = {};
    v8f lacc = {};
#pragma unroll
    for (int r = 0; r < 8; ++r) mrow[r] = -3.0e38f;

    const int nlane = lane & 15;
    const int mofs  = (lane >> 4) << 3;
    const float scale = 0.125f * 1.44269504f;  // 1/sqrt(64) * log2(e)

    const int kr = tid >> 2;            // 0..31
    const int kc = (tid & 3) << 4;
    const int ve = tid >> 1;            // 0..63
    const int vc = (tid & 1) << 4;

    const int nchunk = (qt + 64) / 32;  // causal

    async_b128(&Ks[0][kr][kc],     &khead[(size_t)kr * 64 + kc]);
    async_b128(&Ks[0][kr][kc + 8], &khead[(size_t)kr * 64 + kc + 8]);
    async_b128(&Vs[0][ve][vc],     &vhead[(size_t)ve * 2048 + vc]);
    async_b128(&Vs[0][ve][vc + 8], &vhead[(size_t)ve * 2048 + vc + 8]);

    for (int it = 0; it < nchunk; ++it) {
        const int buf = it & 1;
        const int kt  = it * 32;
        if (it + 1 < nchunk) {
            const int nb = buf ^ 1, kt2 = kt + 32;
            async_b128(&Ks[nb][kr][kc],     &khead[(size_t)(kt2 + kr) * 64 + kc]);
            async_b128(&Ks[nb][kr][kc + 8], &khead[(size_t)(kt2 + kr) * 64 + kc + 8]);
            async_b128(&Vs[nb][ve][vc],     &vhead[(size_t)ve * 2048 + kt2 + vc]);
            async_b128(&Vs[nb][ve][vc + 8], &vhead[(size_t)ve * 2048 + kt2 + vc + 8]);
            S_WAIT_ASYNC("0x4");
        } else {
            S_WAIT_ASYNC("0x0");
        }
        __syncthreads();

        // scores: S[16q x 32k] = Q @ K^T (hd reduced in two 32-chunks)
        v8f sc[2];
#pragma unroll
        for (int nt = 0; nt < 2; ++nt) {
            v8f s = {};
            v16bf kf0 = load_frag(&Ks[buf][nt * 16][0], 72, lane);
            v16bf kf1 = load_frag(&Ks[buf][nt * 16][32], 72, lane);
            s = __builtin_amdgcn_wmma_f32_16x16x32_bf16(false, qf[0], false, kf0,
                                                        (short)0, s, false, false);
            s = __builtin_amdgcn_wmma_f32_16x16x32_bf16(false, qf[1], false, kf1,
                                                        (short)0, s, false, false);
            sc[nt] = s;
        }

        // online softmax in log2 domain; rows live in 16-lane halves
#pragma unroll
        for (int r = 0; r < 8; ++r) {
            int qrow = qt + w * 16 + mofs + r;
            float s0 = sc[0][r] * scale;
            float s1 = sc[1][r] * scale;
            if (kt + nlane > qrow)      s0 = -3.0e38f;
            if (kt + 16 + nlane > qrow) s1 = -3.0e38f;
            float mx = fmaxf(s0, s1);
            mx = fmax_xor_dpp<0x161>(mx);   // ROW_XMASK:1
            mx = fmax_xor_dpp<0x162>(mx);   // ROW_XMASK:2
            mx = fmax_xor_dpp<0x164>(mx);   // ROW_XMASK:4
            mx = fmax_xor_dpp<0x168>(mx);   // ROW_XMASK:8
            float mnew = fmaxf(mrow[r], mx);
            float p0 = exp2f(s0 - mnew);
            float p1 = exp2f(s1 - mnew);
            float corr = exp2f(mrow[r] - mnew);
            mrow[r] = mnew;
            lacc[r] *= corr;
#pragma unroll
            for (int d = 0; d < 4; ++d) oacc[d][r] *= corr;
            Ps[w][mofs + r][nlane]      = (bf16)p0;
            Ps[w][mofs + r][16 + nlane] = (bf16)p1;
        }

        // O += P @ V ; l += P @ ones (row-sum via WMMA, replicated per lane)
        v16bf pf = load_frag(&Ps[w][0][0], 40, lane);
        lacc = __builtin_amdgcn_wmma_f32_16x16x32_bf16(
            false, pf, false, onesf, (short)0, lacc, false, false);
#pragma unroll
        for (int d = 0; d < 4; ++d) {
            v16bf vf = load_frag(&Vs[buf][d * 16][0], 40, lane);
            oacc[d] = __builtin_amdgcn_wmma_f32_16x16x32_bf16(
                false, pf, false, vf, (short)0, oacc[d], false, false);
        }
        __syncthreads();
    }

    // normalize + store [B,S,D] bf16
    int b = bh >> 4, h = bh & 15;
#pragma unroll
    for (int r = 0; r < 8; ++r) {
        float rinv = 1.0f / lacc[r];
        int qrow = qt + w * 16 + mofs + r;
#pragma unroll
        for (int d = 0; d < 4; ++d) {
            float v = oacc[d][r] * rinv;
            ob[((size_t)(b * 2048 + qrow)) * 1024 + h * 64 + d * 16 + nlane] = (bf16)v;
        }
    }
}

// ---------------------------------------------------------------------------
extern "C" void kernel_launch(void* const* d_in, const int* in_sizes, int n_in,
                              void* d_out, int out_size, void* d_ws, size_t ws_size,
                              hipStream_t stream) {
    const float* hs    = (const float*)d_in[0];  // [2,2048,1024]
    const float* attw  = (const float*)d_in[1];  // [1024,3072]
    const float* attb  = (const float*)d_in[2];  // [3072]
    const float* projw = (const float*)d_in[3];  // [1024,1024]
    const float* projb = (const float*)d_in[4];  // [1024]

    char* ws = (char*)d_ws;
    bf16* hs_bf  = (bf16*)ws; ws += (size_t)4096 * 1024 * 2;
    bf16* attwT  = (bf16*)ws; ws += (size_t)3072 * 1024 * 2;
    bf16* projwT = (bf16*)ws; ws += (size_t)1024 * 1024 * 2;
    bf16* qb  = (bf16*)ws;    ws += (size_t)2 * 16 * 2048 * 64 * 2;
    bf16* kb  = (bf16*)ws;    ws += (size_t)2 * 16 * 2048 * 64 * 2;
    bf16* vTb = (bf16*)ws;    ws += (size_t)2 * 16 * 2048 * 64 * 2;
    bf16* ob  = (bf16*)ws;    ws += (size_t)4096 * 1024 * 2;

    cvt4_f32_bf16<<<(4096 * 1024 / 4) / 256, 256, 0, stream>>>(hs, hs_bf);
    cvt_transpose<<<dim3(3072 / 32, 1024 / 32), 256, 0, stream>>>(attw, attwT, 1024, 3072);
    cvt_transpose<<<dim3(1024 / 32, 1024 / 32), 256, 0, stream>>>(projw, projwT, 1024, 1024);

    gemm_bf16<0><<<dim3(3072 / 128, 4096 / 256), 256, 0, stream>>>(
        hs_bf, attwT, attb, 3072, qb, kb, vTb, nullptr);
    attn_kernel<<<dim3(32, 32), 128, 0, stream>>>(qb, kb, vTb, ob);
    gemm_bf16<1><<<dim3(1024 / 128, 4096 / 256), 256, 0, stream>>>(
        ob, projwT, projb, 1024, nullptr, nullptr, nullptr, (float*)d_out);
}
